// qconv2d_8bit_50629074485863
// MI455X (gfx1250) — 8-bit matrix operand order repaired and checked by reading, compiled, not run on hardware
//
#include <hip/hip_runtime.h>
#include <cstdint>

typedef int v8i __attribute__((ext_vector_type(8)));

namespace {
constexpr int B_ = 16, CIN_ = 128, H_ = 96, W_ = 96, COUT_ = 128;
constexpr int HP_ = 98, WP_ = 98;
constexpr size_t XPAD_BYTES = (size_t)B_ * HP_ * WP_ * CIN_;
constexpr int WELEMS = COUT_ * 9 * CIN_;
constexpr int LDS_STRIDE = 144;
constexpr int WROW = 9 * CIN_;
constexpr int WROW_LDS = 1168;
constexpr int WCHUNKS = WELEMS / 16;
constexpr unsigned WLDS_BYTES = (unsigned)COUT_ * WROW_LDS;
}

__global__ __launch_bounds__(256) void qconv_border_fill(int8_t* __restrict__ xpad,
                                                         const float* __restrict__ p_zp) {
  constexpr int TOTAL = B_ * 388 * 8;
  int idx = blockIdx.x * 256 + threadIdx.x;
  if (idx >= TOTAL) return;
  int chunk = idx & 7;
  int t = idx >> 3;
  int b = t / 388;
  int p = t % 388;
  int hp, wp;
  if (p < 98)       { hp = 0;           wp = p;       }
  else if (p < 196) { hp = HP_ - 1;     wp = p - 98;  }
  else if (p < 292) { hp = p - 196 + 1; wp = 0;       }
  else              { hp = p - 292 + 1; wp = WP_ - 1; }
  int zi = (int)rintf(p_zp[0]);
  int w  = (zi & 0xFF) * 0x01010101;
  size_t off = (((size_t)b * HP_ + hp) * WP_ + wp) * CIN_ + (size_t)chunk * 16;
  *(int4*)(xpad + off) = make_int4(w, w, w, w);
}

__global__ __launch_bounds__(256) void qconv_quantize(const float* __restrict__ x,
                                                      int8_t* __restrict__ xpad,
                                                      const float* __restrict__ p_delta,
                                                      const float* __restrict__ p_zp) {
  __shared__ int8_t tile[W_ * LDS_STRIDE];
  const float inv = 1.0f / p_delta[0];
  const float zp  = p_zp[0];
  const int bh = blockIdx.x;
  const int b  = bh / H_;
  const int h  = bh % H_;
  const float* xrow = x + (size_t)b * CIN_ * H_ * W_ + (size_t)h * W_;

  for (int i = threadIdx.x; i < CIN_ * W_; i += 256) {
    int ci = i / W_;
    int w  = i % W_;
    float q = rintf(xrow[(size_t)ci * (H_ * W_) + w] * inv) + zp;
    q = fminf(fmaxf(q, -128.f), 127.f);
    tile[w * LDS_STRIDE + ci] = (int8_t)(int)q;
  }
  __syncthreads();

  int8_t* orow = xpad + (((size_t)b * HP_ + (h + 1)) * WP_ + 1) * CIN_;
  for (int i = threadIdx.x; i < (W_ * CIN_) / 16; i += 256) {
    int w = i >> 3, part = i & 7;
    *(int4*)(orow + (size_t)i * 16) = *(const int4*)(tile + w * LDS_STRIDE + part * 16);
  }
}

__global__ __launch_bounds__(256) void qconv_pack_w(const int* __restrict__ w32,
                                                    int8_t* __restrict__ w8) {
  int i = blockIdx.x * 256 + threadIdx.x;
  if (i * 4 >= WELEMS) return;
  int4 v = ((const int4*)w32)[i];
  *(char4*)(w8 + (size_t)i * 4) =
      make_char4((char)v.x, (char)v.y, (char)v.z, (char)v.w);
}

__global__ __launch_bounds__(256) void qconv_wmma(const int8_t* __restrict__ xpad,
                                                  const int8_t* __restrict__ w8,
                                                  const float* __restrict__ bias,
                                                  const float* __restrict__ zpsum,
                                                  const float* __restrict__ scale,
                                                  float* __restrict__ out) {
  extern __shared__ int8_t bsm[];

#pragma unroll 4
  for (int k = 0; k < WCHUNKS / 256; ++k) {
    int j = threadIdx.x + k * 256;
    int row = j / (WROW / 16);
    int part = j % (WROW / 16);
    unsigned ldsoff = (unsigned)row * WROW_LDS + (unsigned)part * 16;
    unsigned long long gaddr = (unsigned long long)(uintptr_t)(w8 + (size_t)j * 16);
    asm volatile("global_load_async_to_lds_b128 %0, %1, off"
                 :: "v"(ldsoff), "v"(gaddr)
                 : "memory");
  }
  asm volatile("s_wait_asynccnt 0" ::: "memory");
  __syncthreads();

  const int lane = threadIdx.x & 31;
  const int wave = threadIdx.x >> 5;
  const int l = lane & 15;
  const int h = lane >> 4;

  const int tileM  = blockIdx.x * 8 + wave;
  const int pixel0 = tileM * 16;
  const int b   = pixel0 / (H_ * W_);
  const int rem = pixel0 % (H_ * W_);
  const int ho  = rem / W_;
  const int wo  = rem % W_;

  v8i acc[8];
  const v8i vzero = {0, 0, 0, 0, 0, 0, 0, 0};
#pragma unroll
  for (int i = 0; i < 8; ++i) acc[i] = vzero;

  const int8_t* arow0 = xpad + (((size_t)b * HP_ + ho) * WP_ + (wo + l)) * CIN_;

#pragma unroll
  for (int kh = 0; kh < 3; ++kh) {
#pragma unroll
    for (int kw = 0; kw < 3; ++kw) {
      const int kpos = kh * 3 + kw;
      const int8_t* arow = arow0 + ((size_t)kh * WP_ + kw) * CIN_;
#pragma unroll
      for (int c = 0; c < 2; ++c) {
        const int8_t* ab = arow + c * 64 + h * 16;
        int2 a0 = *(const int2*)(ab + 0);
        int2 a1 = *(const int2*)(ab + 8);
        int2 a2 = *(const int2*)(ab + 32);
        int2 a3 = *(const int2*)(ab + 40);
        v8i A = {a0.x, a0.y, a1.x, a1.y, a2.x, a2.y, a3.x, a3.y};
#pragma unroll
        for (int nt = 0; nt < 8; ++nt) {
          const int8_t* wb = bsm + (nt * 16 + l) * WROW_LDS
                                 + kpos * CIN_ + c * 64 + h * 16;
          int4 b0 = *(const int4*)(wb);
          int4 b1 = *(const int4*)(wb + 32);
          v8i Bf = {b0.x, b0.y, b0.z, b0.w, b1.x, b1.y, b1.z, b1.w};
          acc[nt] = __builtin_amdgcn_wmma_i32_16x16x64_iu8(
               true, A,  true, Bf, acc[nt],
               false,  false);
        }
      }
    }
  }

#pragma unroll
  for (int nt = 0; nt < 8; ++nt) {
    const int n = nt * 16 + l;
    const float zs = zpsum[n];
    const float sc = scale[n];
    const float bs = bias[n];
    float* obase = out + (((size_t)b * COUT_ + n) * H_ + ho) * W_ + wo + 8 * h;
    float v[8];
#pragma unroll
    for (int r = 0; r < 8; ++r)
      v[r] = (float)(_Float16)(((float)acc[nt][r] - zs) * sc + bs);
    *(float4*)(obase + 0) = make_float4(v[0], v[1], v[2], v[3]);
    *(float4*)(obase + 4) = make_float4(v[4], v[5], v[6], v[7]);
  }
}

extern "C" void kernel_launch(void* const* d_in, const int* in_sizes, int n_in,
                              void* d_out, int out_size, void* d_ws, size_t ws_size,
                              hipStream_t stream) {
  const float* x     = (const float*)d_in[0];
  const int*   w32   = (const int*)d_in[1];
  const float* bias  = (const float*)d_in[2];
  const float* delta = (const float*)d_in[3];
  const float* zp    = (const float*)d_in[4];
  const float* zpsum = (const float*)d_in[5];
  const float* scale = (const float*)d_in[6];
  float* out = (float*)d_out;

  int8_t* xpad = (int8_t*)d_ws;
  int8_t* w8   = (int8_t*)d_ws + XPAD_BYTES;

  qconv_border_fill<<<(B_ * 388 * 8 + 255) / 256, 256, 0, stream>>>(xpad, zp);
  qconv_quantize<<<B_ * H_, 256, 0, stream>>>(x, xpad, delta, zp);
  qconv_pack_w<<<(WELEMS / 4 + 255) / 256, 256, 0, stream>>>(w32, w8);
  qconv_wmma<<<(B_ * H_ * W_) / 128, 256, WLDS_BYTES, stream>>>(xpad, w8, bias,
                                                                zpsum, scale, out);
}
